// PointMLPBlock_37933151158903
// MI455X (gfx1250) — matrix operand order repaired and checked by reading, not compiled, not run on hardware
//
#include <hip/hip_runtime.h>
#include <hip/hip_bf16.h>

#define BB   4
#define NN   4096
#define KK   24
#define CIN  64
#define C2   128      // 2*CIN = GEMM inner dim
#define COUT 128
#define EPSF 1e-5f

typedef __attribute__((ext_vector_type(8)))  _Float16 v8h;
typedef __attribute__((ext_vector_type(16))) _Float16 v16h;
typedef __attribute__((ext_vector_type(8)))  float    v8f;

// ---------------------------------------------------------------------------
// gfx1250 async global->LDS copy (ASYNCcnt-tracked), via inline asm so it is
// portable across toolchain builtin-arity differences.
// ---------------------------------------------------------------------------
__device__ __forceinline__ void async_copy_b128(unsigned lds_off, const void* gptr)
{
    asm volatile("global_load_async_to_lds_b128 %0, %1, off"
                 :: "v"(lds_off), "v"((unsigned long long)(uintptr_t)gptr)
                 : "memory");
}
__device__ __forceinline__ void async_wait0()
{
    asm volatile("s_wait_asynccnt 0" ::: "memory");
}

// ---------------------------------------------------------------------------
// Shared WMMA GEMM core: Y[128, 64cols] = W[128x128] * Xtile[128 x 64] + bias
// Wl: LDS, row-major [outCh][k] f16.  Xl: LDS, [col][k] f16 (k fastest).
// Per-channel sum/sumsq accumulated in registers across column tiles, then
// flushed once via LDS atomics.
// ---------------------------------------------------------------------------
__device__ __forceinline__ void gemm_core(const _Float16* Wl, const _Float16* Xl,
                                          float* s_sum, float* s_sq,
                                          const float* __restrict__ bias,
                                          _Float16* __restrict__ Y, size_t cbase)
{
    const int tid   = threadIdx.x;
    const int wave  = tid >> 5;
    const int lane  = tid & 31;
    const int l15   = lane & 15;
    const bool hi   = (lane & 16) != 0;
    const int mbase = wave * 16;              // 8 waves x 16 rows = 128 channels
    const int chb   = mbase + (hi ? 8 : 0);   // D layout: VGPR r -> M = 8*hi + r

    float bv[8], psum[8], psq[8];
    #pragma unroll
    for (int r = 0; r < 8; ++r) { bv[r] = bias[chb + r]; psum[r] = 0.f; psq[r] = 0.f; }

    for (int ct = 0; ct < 4; ++ct) {          // 4 x 16 = 64 columns
        v8f acc = {};
        #pragma unroll
        for (int ks = 0; ks < 4; ++ks) {      // 4 x 32 = 128 inner dim
            const int kb = ks * 32;
            // A fragment 16x32 f16: lane holds row mbase+l15,
            // low half-wave K{kb+0..7, kb+16..23}, high half-wave K{kb+8..15, kb+24..31}
            const _Float16* wrow = Wl + (size_t)(mbase + l15) * C2 + kb + (hi ? 8 : 0);
            v8h alo = *(const v8h*)(wrow);
            v8h ahi = *(const v8h*)(wrow + 16);
            v16h a  = __builtin_shufflevector(alo, ahi, 0,1,2,3,4,5,6,7,8,9,10,11,12,13,14,15);
            // B fragment 32x16 f16: lane holds column ct*16+l15,
            // low half-wave K{kb+0..7, kb+16..23}, high half-wave K{kb+8..15, kb+24..31}
            const _Float16* xcol = Xl + (size_t)(ct * 16 + l15) * C2 + kb + (hi ? 8 : 0);
            v8h blo = *(const v8h*)(xcol);
            v8h bhi = *(const v8h*)(xcol + 16);
            v16h b  = __builtin_shufflevector(blo, bhi, 0,1,2,3,4,5,6,7,8,9,10,11,12,13,14,15);
            acc = __builtin_amdgcn_wmma_f32_16x16x32_f16(false, a, false, b,
                                                         (short)0, acc, false, false);
        }
        const size_t col = cbase + ct * 16 + l15;
        v8h ov;
        #pragma unroll
        for (int r = 0; r < 8; ++r) {
            const float y = acc[r] + bv[r];
            psum[r] += y; psq[r] += y * y;
            ov[r] = (_Float16)y;
        }
        *(v8h*)(Y + col * COUT + chb) = ov;   // chb multiple of 8 -> 16B aligned
    }
    #pragma unroll
    for (int r = 0; r < 8; ++r) {
        atomicAdd(&s_sum[chb + r], psum[r]);
        atomicAdd(&s_sq[chb + r],  psq[r]);
    }
}

// ---------------------------------------------------------------------------
// Generic GEMM stage: weights (pre-converted f16) staged via async LDS copy;
// optional BN-fold (relu(x*sc+sh)) applied on vectorized X load.
// ---------------------------------------------------------------------------
__global__ void k_gemm(const _Float16* __restrict__ X,
                       const float* __restrict__ fsc, const float* __restrict__ fsh,
                       int fold,
                       const _Float16* __restrict__ Wh, const float* __restrict__ bias,
                       _Float16* __restrict__ Y,
                       float* __restrict__ sum_out, float* __restrict__ sq_out)
{
    __shared__ __align__(16) _Float16 Wl[COUT * C2];
    __shared__ __align__(16) _Float16 Xl[64 * C2];
    __shared__ float s_sum[COUT], s_sq[COUT];
    const int tid = threadIdx.x;

    // async-stage 32KB of f16 weights: 2048 x b128, 8 per thread
    {
        const unsigned wl0 = (unsigned)(uintptr_t)(void*)&Wl[0];
        const char* g = (const char*)Wh;
        for (int c = tid; c < 2048; c += 256)
            async_copy_b128(wl0 + c * 16, g + c * 16);
    }
    if (tid < COUT) { s_sum[tid] = 0.f; s_sq[tid] = 0.f; }

    const size_t cbase = (size_t)blockIdx.x * 64;
    for (int i = tid; i < 64 * 16; i += 256) {     // 16 v8h chunks per column
        v8h v = *(const v8h*)(X + cbase * C2 + (size_t)i * 8);
        if (fold) {
            const int ch0 = (i & 15) * 8;
            #pragma unroll
            for (int e = 0; e < 8; ++e) {
                const float f = (float)v[e] * fsc[ch0 + e] + fsh[ch0 + e];
                v[e] = (_Float16)fmaxf(f, 0.f);
            }
        }
        *(v8h*)(Xl + (size_t)i * 8) = v;
    }
    async_wait0();
    __syncthreads();
    gemm_core(Wl, Xl, s_sum, s_sq, bias, Y, cbase);
    __syncthreads();
    for (int i = tid; i < COUT; i += 256) {
        atomicAdd(&sum_out[i], s_sum[i]);
        atomicAdd(&sq_out[i],  s_sq[i]);
    }
}

// ---------------------------------------------------------------------------
// Transfer GEMM: builds newp columns (normalized neighbor diff ++ anchor) in
// LDS on the fly from gather indices; weights async-staged; same WMMA core.
// ---------------------------------------------------------------------------
__global__ void k_gemm_transfer(const float* __restrict__ x,
                                const int* __restrict__ idx,
                                const float* __restrict__ inv_std,
                                const float* __restrict__ alpha,
                                const float* __restrict__ beta,
                                const _Float16* __restrict__ Wh,
                                const float* __restrict__ bias,
                                _Float16* __restrict__ Y,
                                float* __restrict__ sum_out, float* __restrict__ sq_out)
{
    __shared__ __align__(16) _Float16 Wl[COUT * C2];
    __shared__ __align__(16) _Float16 Xl[64 * C2];
    __shared__ float s_sum[COUT], s_sq[COUT];
    const int tid = threadIdx.x;

    {
        const unsigned wl0 = (unsigned)(uintptr_t)(void*)&Wl[0];
        const char* g = (const char*)Wh;
        for (int c = tid; c < 2048; c += 256)
            async_copy_b128(wl0 + c * 16, g + c * 16);
    }
    if (tid < COUT) { s_sum[tid] = 0.f; s_sq[tid] = 0.f; }

    const size_t cbase = (size_t)blockIdx.x * 64;
    {   // 4 threads per column, 32 channels each
        const int col = tid >> 2, p = tid & 3;
        const size_t c = cbase + col;
        const int kk = (int)(c % KK);
        const size_t t = c / KK;
        const int n = (int)(t & (NN - 1));
        const int b = (int)(t >> 12);
        const int m = idx[((size_t)b * NN + n) * KK + kk];
        const float inv = inv_std[b];
        const float* xb = x + (size_t)b * CIN * NN;
        if (p < 2) {
            for (int c0 = p * 32; c0 < p * 32 + 32; ++c0) {
                const float d = xb[(size_t)c0 * NN + m] - xb[(size_t)c0 * NN + n];
                Xl[col * C2 + c0] = (_Float16)(alpha[c0] * (d * inv) + beta[c0]);
            }
        } else {
            for (int c0 = (p - 2) * 32; c0 < (p - 2) * 32 + 32; ++c0)
                Xl[col * C2 + CIN + c0] = (_Float16)xb[(size_t)c0 * NN + n];
        }
    }
    async_wait0();
    __syncthreads();
    gemm_core(Wl, Xl, s_sum, s_sq, bias, Y, cbase);
    __syncthreads();
    for (int i = tid; i < COUT; i += 256) {
        atomicAdd(&sum_out[i], s_sum[i]);
        atomicAdd(&sq_out[i],  s_sq[i]);
    }
}

// ---------------------------------------------------------------------------
// Convert the five 128x128 f32 weight matrices to f16 (enables raw async copy)
// ---------------------------------------------------------------------------
__global__ void k_prep_w(const float* __restrict__ w0, const float* __restrict__ w1,
                         const float* __restrict__ w2, const float* __restrict__ w3,
                         const float* __restrict__ w4, _Float16* __restrict__ out)
{
    const int i = blockIdx.x * 256 + threadIdx.x;      // 0 .. 5*16384-1
    const int m = i >> 14, r = i & 16383;
    const float* src = (m == 0) ? w0 : (m == 1) ? w1 : (m == 2) ? w2 : (m == 3) ? w3 : w4;
    out[i] = (_Float16)src[r];
}

// ---------------------------------------------------------------------------
// kNN: 128 points per block; tile xyz through LDS; per-point insertion-sorted
// top-K list kept in LDS (stable ascending -> matches top_k tie-break).
// ---------------------------------------------------------------------------
__global__ void k_knn(const float* __restrict__ xyz, int* __restrict__ idx)
{
    __shared__ float px[128], py[128], pz[128], sq[128];
    __shared__ float topd[128 * KK];
    __shared__ int   topi[128 * KK];
    const int tid = threadIdx.x;
    const int b   = blockIdx.x >> 5;                 // 32 blocks / batch
    const int n   = ((blockIdx.x & 31) << 7) + tid;
    const float* xb = xyz + (size_t)b * 3 * NN;
    const float ax = xb[n], ay = xb[NN + n], az = xb[2 * NN + n];
    const float asq = ax * ax + ay * ay + az * az;
    float* td = topd + tid * KK;
    int*   ti = topi + tid * KK;
    int cnt = 0;
    for (int mt = 0; mt < NN; mt += 128) {
        const float vx = xb[mt + tid], vy = xb[NN + mt + tid], vz = xb[2 * NN + mt + tid];
        px[tid] = vx; py[tid] = vy; pz[tid] = vz; sq[tid] = vx * vx + vy * vy + vz * vz;
        __syncthreads();
        for (int j = 0; j < 128; ++j) {
            const float d = asq + sq[j] - 2.0f * (ax * px[j] + ay * py[j] + az * pz[j]);
            int pos;
            if (cnt < KK)            { pos = cnt++; }
            else if (d < td[KK - 1]) { pos = KK - 1; }
            else continue;
            while (pos > 0 && td[pos - 1] > d) {
                td[pos] = td[pos - 1]; ti[pos] = ti[pos - 1]; --pos;
            }
            td[pos] = d; ti[pos] = mt + j;
        }
        __syncthreads();
    }
    int* op = idx + ((size_t)b * NN + n) * KK;
    for (int k2 = 0; k2 < KK; ++k2) op[k2] = ti[k2];
}

// ---------------------------------------------------------------------------
// Per-batch sum / sumsq of neighbor diffs (for anchor-norm std, ddof=1).
// ---------------------------------------------------------------------------
__global__ void k_diffstats(const float* __restrict__ x, const int* __restrict__ idx,
                            float* __restrict__ acc /* [B][2] */)
{
    __shared__ float rs[256], rss[256];
    const int tid = threadIdx.x;
    const int b = blockIdx.x >> 4;                   // 16 blocks / batch
    const int n = ((blockIdx.x & 15) << 8) + tid;
    const float* xb = x + (size_t)b * CIN * NN;
    const int* ip = idx + ((size_t)b * NN + n) * KK;
    float s = 0.f, ss = 0.f;
    for (int kk = 0; kk < KK; ++kk) {
        const int m = ip[kk];
        for (int c = 0; c < CIN; ++c) {
            const float d = xb[(size_t)c * NN + m] - xb[(size_t)c * NN + n];
            s += d; ss += d * d;
        }
    }
    rs[tid] = s; rss[tid] = ss;
    __syncthreads();
    for (int off = 128; off > 0; off >>= 1) {
        if (tid < off) { rs[tid] += rs[tid + off]; rss[tid] += rss[tid + off]; }
        __syncthreads();
    }
    if (tid == 0) { atomicAdd(&acc[b * 2], rs[0]); atomicAdd(&acc[b * 2 + 1], rss[0]); }
}

__global__ void k_finalize_std(const float* __restrict__ acc, float* __restrict__ inv_std)
{
    const int b = threadIdx.x;
    if (b < BB) {
        const double M = (double)NN * KK * CIN;
        const double s = acc[b * 2], ss = acc[b * 2 + 1];
        const float var = (float)((ss - s * s / M) / (M - 1.0));
        inv_std[b] = 1.0f / (sqrtf(fmaxf(var, 0.f)) + EPSF);
    }
}

__global__ void k_finalize_bn(const float* __restrict__ sum, const float* __restrict__ sq,
                              const float* __restrict__ g, const float* __restrict__ bt,
                              float* __restrict__ sc, float* __restrict__ sh, float cntInv)
{
    const int c = threadIdx.x;
    if (c < COUT) {
        const float mean = sum[c] * cntInv;
        const float var  = sq[c] * cntInv - mean * mean;
        const float s    = g[c] * rsqrtf(fmaxf(var, 0.f) + EPSF);
        sc[c] = s;
        sh[c] = bt[c] - mean * s;
    }
}

// relu(bn3(y3) + relu(bn1(y1))) then max over K neighbors
__global__ void k_pool(const _Float16* __restrict__ y1, const _Float16* __restrict__ y3,
                       const float* __restrict__ sc1, const float* __restrict__ sh1,
                       const float* __restrict__ sc3, const float* __restrict__ sh3,
                       _Float16* __restrict__ ph, float* __restrict__ pf)
{
    const size_t gid  = (size_t)blockIdx.x * 256 + threadIdx.x;
    const size_t col2 = gid >> 7;
    const int    ch   = (int)(gid & 127);
    const float a1 = sc1[ch], c1 = sh1[ch], a3 = sc3[ch], c3 = sh3[ch];
    const size_t base = col2 * KK * COUT + ch;
    float m = 0.f;                                   // relu output >= 0
    for (int kk = 0; kk < KK; ++kk) {
        const float v1 = fmaxf((float)y1[base + (size_t)kk * COUT] * a1 + c1, 0.f);
        const float v3 = (float)y3[base + (size_t)kk * COUT] * a3 + c3;
        m = fmaxf(m, fmaxf(v3 + v1, 0.f));
    }
    ph[col2 * COUT + ch] = (_Float16)m;
    pf[col2 * COUT + ch] = m;
}

// out[b, ch, n] = relu(bn5(y5) + pooled)
__global__ void k_final(const _Float16* __restrict__ y5,
                        const float* __restrict__ sc5, const float* __restrict__ sh5,
                        const float* __restrict__ pf, float* __restrict__ out)
{
    const size_t gid  = (size_t)blockIdx.x * 256 + threadIdx.x;
    const size_t col2 = gid >> 7;
    const int    ch   = (int)(gid & 127);
    float v = (float)y5[col2 * COUT + ch] * sc5[ch] + sh5[ch] + pf[col2 * COUT + ch];
    v = fmaxf(v, 0.f);
    const size_t b = col2 >> 12, n = col2 & (NN - 1);
    out[(b * COUT + ch) * NN + n] = v;
}

__global__ void k_zero(float* __restrict__ p, int n)
{
    const int i = blockIdx.x * 256 + threadIdx.x;
    if (i < n) p[i] = 0.f;
}

// ---------------------------------------------------------------------------
extern "C" void kernel_launch(void* const* d_in, const int* in_sizes, int n_in,
                              void* d_out, int out_size, void* d_ws, size_t ws_size,
                              hipStream_t stream)
{
    const float* x     = (const float*)d_in[0];
    const float* xyz   = (const float*)d_in[1];
    const float* alpha = (const float*)d_in[2];
    const float* beta  = (const float*)d_in[3];
    const float* wt    = (const float*)d_in[4];
    const float* bt_   = (const float*)d_in[5];
    const float* gt    = (const float*)d_in[6];
    const float* btt   = (const float*)d_in[7];
    const float* Wrb[4]  = { (const float*)d_in[8],  (const float*)d_in[12],
                             (const float*)d_in[16], (const float*)d_in[20] };
    const float* Brb[4]  = { (const float*)d_in[9],  (const float*)d_in[13],
                             (const float*)d_in[17], (const float*)d_in[21] };
    const float* Grb[4]  = { (const float*)d_in[10], (const float*)d_in[14],
                             (const float*)d_in[18], (const float*)d_in[22] };
    const float* BTrb[4] = { (const float*)d_in[11], (const float*)d_in[15],
                             (const float*)d_in[19], (const float*)d_in[23] };
    float* out = (float*)d_out;

    // ---- workspace layout (bump allocator, 256B aligned) ----
    char* ws = (char*)d_ws;
    size_t off = 0;
    auto alloc = [&](size_t bytes) -> char* {
        char* p = ws + off;
        off = (off + bytes + 255) & ~(size_t)255;
        return p;
    };
    const size_t colsBig   = (size_t)BB * NN * KK;   // 393216
    const size_t colsSmall = (size_t)BB * NN;        // 16384

    int*    idx    = (int*)alloc(colsBig * sizeof(int));
    // contiguous float stats block: [0..7] batch_acc | [8..647] st_sum(5x128)
    // | [648..1287] st_sq | [1288..1291] inv_std | [1292..1931] sc | [1932..2571] sh
    float*  stats  = (float*)alloc(2572 * sizeof(float));
    float*  batch_acc = stats;
    float*  st_sum  = stats + 8;
    float*  st_sq   = stats + 8 + 5 * 128;
    float*  inv_std = stats + 1288;
    float*  sc      = stats + 1292;
    float*  sh      = stats + 1932;
    _Float16* Wh   = (_Float16*)alloc(5 * COUT * C2 * sizeof(_Float16));    // 160 KB f16
    _Float16* y1   = (_Float16*)alloc(colsBig * COUT * sizeof(_Float16));   // ~100 MB
    _Float16* y2   = (_Float16*)alloc(colsBig * COUT * sizeof(_Float16));   // ~100 MB
    _Float16* y3   = y2;                                                    // in-place
    _Float16* y4   = (_Float16*)alloc(colsSmall * COUT * sizeof(_Float16));
    _Float16* y5   = (_Float16*)alloc(colsSmall * COUT * sizeof(_Float16));
    _Float16* ph   = (_Float16*)alloc(colsSmall * COUT * sizeof(_Float16));
    float*    pf   = (float*)alloc(colsSmall * COUT * sizeof(float));
    (void)ws_size; (void)in_sizes; (void)n_in; (void)out_size;

    const int gBig   = (int)(colsBig / 64);            // 6144
    const int gSmall = (int)(colsSmall / 64);          // 256
    const int gElem  = (int)(colsSmall * COUT / 256);  // 8192
    const float invBig   = 1.0f / (float)(colsBig);
    const float invSmall = 1.0f / (float)(colsSmall);

    // Stage 0: zero stat accumulators; convert weights to f16 for async staging
    k_zero<<<(1288 + 255) / 256, 256, 0, stream>>>(stats, 1288);
    k_prep_w<<<5 * 16384 / 256, 256, 0, stream>>>(wt, Wrb[0], Wrb[1], Wrb[2], Wrb[3], Wh);
    // Stage 1: kNN indices
    k_knn<<<BB * (NN / 128), 128, 0, stream>>>(xyz, idx);
    // Stage 2: per-batch std of diffs
    k_diffstats<<<BB * (NN / 256), 256, 0, stream>>>(x, idx, batch_acc);
    k_finalize_std<<<1, 32, 0, stream>>>(batch_acc, inv_std);
    // Stage 3: transfer conv (gather+normalize+concat fused into GEMM load)
    k_gemm_transfer<<<gBig, 256, 0, stream>>>(x, idx, inv_std, alpha, beta,
                                              Wh + 0 * 16384, bt_, y1,
                                              st_sum + 0 * 128, st_sq + 0 * 128);
    k_finalize_bn<<<1, 128, 0, stream>>>(st_sum + 0 * 128, st_sq + 0 * 128, gt, btt,
                                         sc + 0 * 128, sh + 0 * 128, invBig);
    // Stage 4: res-block conv 1 (fold bn0+relu on load)
    k_gemm<<<gBig, 256, 0, stream>>>(y1, sc + 0 * 128, sh + 0 * 128, 1,
                                     Wh + 1 * 16384, Brb[0], y2,
                                     st_sum + 1 * 128, st_sq + 1 * 128);
    k_finalize_bn<<<1, 128, 0, stream>>>(st_sum + 1 * 128, st_sq + 1 * 128, Grb[0], BTrb[0],
                                         sc + 1 * 128, sh + 1 * 128, invBig);
    // Stage 5: res-block conv 2 (in-place y2 -> y3)
    k_gemm<<<gBig, 256, 0, stream>>>(y2, sc + 1 * 128, sh + 1 * 128, 1,
                                     Wh + 2 * 16384, Brb[1], y3,
                                     st_sum + 2 * 128, st_sq + 2 * 128);
    k_finalize_bn<<<1, 128, 0, stream>>>(st_sum + 2 * 128, st_sq + 2 * 128, Grb[1], BTrb[1],
                                         sc + 2 * 128, sh + 2 * 128, invBig);
    // Stage 6: residual + relu + max-pool over K
    k_pool<<<gElem, 256, 0, stream>>>(y1, y3, sc + 0 * 128, sh + 0 * 128,
                                      sc + 2 * 128, sh + 2 * 128, ph, pf);
    // Stage 7: second res-block conv 1 (no fold; pooled already activated)
    k_gemm<<<gSmall, 256, 0, stream>>>(ph, sc, sh, 0,
                                       Wh + 3 * 16384, Brb[2], y4,
                                       st_sum + 3 * 128, st_sq + 3 * 128);
    k_finalize_bn<<<1, 128, 0, stream>>>(st_sum + 3 * 128, st_sq + 3 * 128, Grb[2], BTrb[2],
                                         sc + 3 * 128, sh + 3 * 128, invSmall);
    // Stage 8: second res-block conv 2
    k_gemm<<<gSmall, 256, 0, stream>>>(y4, sc + 3 * 128, sh + 3 * 128, 1,
                                       Wh + 4 * 16384, Brb[3], y5,
                                       st_sum + 4 * 128, st_sq + 4 * 128);
    k_finalize_bn<<<1, 128, 0, stream>>>(st_sum + 4 * 128, st_sq + 4 * 128, Grb[3], BTrb[3],
                                         sc + 4 * 128, sh + 4 * 128, invSmall);
    // Stage 9: bn + residual + relu, write [B, COUT, N]
    k_final<<<gElem, 256, 0, stream>>>(y5, sc + 4 * 128, sh + 4 * 128, pf, out);
}
